// SHCA_54709293416667
// MI455X (gfx1250) — matrix operand order repaired and checked by reading, not compiled, not run on hardware
//
#include <hip/hip_runtime.h>
#include <hip/hip_bf16.h>

typedef __bf16 bf16;
typedef __attribute__((ext_vector_type(16))) __bf16 bf16x16;
typedef __attribute__((ext_vector_type(8)))  __bf16 bf16x8;
typedef __attribute__((ext_vector_type(8)))  float  f32x8;

#define XLEN 2048
#define YLEN 1024
#define BB   8
#define XD   1024
#define YD   1024
#define HD   512
#define YOUT 1024
#define CD   (YD + HD)   // 1536
#define MBX  (XLEN * BB) // 16384 rows of X
#define MBY  (YLEN * BB) // 8192 rows of Y
#define FMAXF 3.402823466e38f

// ---------------- fragment loads (CDNA5 WMMA bf16 16x16x32 layouts) ------------
// A 16x32: lane<16 -> M=lane, K=0..15; lane>=16 -> M=lane-16, K=16..31 (contiguous)
__device__ __forceinline__ bf16x16 load_frag_a(const bf16* p) {
    union { bf16x16 v; bf16x8 h[2]; } u;
    u.h[0] = *(const bf16x8*)(p);
    u.h[1] = *(const bf16x8*)(p + 8);
    return u.v;
}
// B 32x16: lane&15 = column, lane<16 -> K=0..15, lane>=16 -> K=16..31 (contiguous)
__device__ __forceinline__ bf16x16 load_frag_b(const bf16* p) {
    union { bf16x16 v; bf16x8 h[2]; } u;
    u.h[0] = *(const bf16x8*)(p);
    u.h[1] = *(const bf16x8*)(p + 8);
    return u.v;
}

// ---------------- generic WMMA GEMM: C[m,n] = sum_k A[m,k] * Bt[n,k] ------------
// Each wave computes a 32(M) x 64(N) tile: 2 A-frags, 4 shared B-frags, 8 WMMAs / K-step.
// MODE 0: bf16 out, optional bias_n / bias_m
// MODE 1: f32 out, bias_n
// MODE 2: f32 logits epilogue (scale, gated alignment bias, clamp, pad mask)
template <int MODE>
__global__ __launch_bounds__(256) void gemm_wmma(
    const bf16* __restrict__ A,  int lda, long long a_batch,
    const bf16* __restrict__ Bt, int ldb, long long bt_batch,
    const float* __restrict__ bias_n, const float* __restrict__ bias_m,
    void* __restrict__ Cout, long long c_base, long long c_batch, int c_rs,
    int M, int N, int K,
    const float* __restrict__ align_bias, const unsigned char* __restrict__ mask,
    const float* __restrict__ gate_w, float scale)
{
    const int lane = threadIdx.x & 31;
    const int wid  = threadIdx.x >> 5;
    const int l    = lane & 15;
    const int hi   = lane >> 4;
    const int tilesN = N >> 6;             // 64 cols per wave
    const int tilesM = M >> 5;             // 32 rows per wave
    const int tile = blockIdx.x * 8 + wid; // wave-uniform
    if (tile >= tilesM * tilesN) return;   // whole wave exits -> EXEC stays all-ones
    const int tm = tile / tilesN;
    const int tn = tile % tilesN;
    const int z  = blockIdx.z;
    const bf16* Ab = A  + (long long)z * a_batch;
    const bf16* Bb = Bt + (long long)z * bt_batch;
    const int m0 = tm * 32;
    const int n0 = tn * 64;

    // per-lane base pointers
    const bf16* ap0 = Ab + (size_t)(m0 + l) * lda + hi * 16;
    const bf16* ap1 = ap0 + (size_t)16 * lda;
    const bf16* bp0 = Bb + (size_t)(n0 +  0 + l) * ldb + hi * 16;
    const bf16* bp1 = Bb + (size_t)(n0 + 16 + l) * ldb + hi * 16;
    const bf16* bp2 = Bb + (size_t)(n0 + 32 + l) * ldb + hi * 16;
    const bf16* bp3 = Bb + (size_t)(n0 + 48 + l) * ldb + hi * 16;

    f32x8 acc[2][4] = {};
    for (int k0 = 0; k0 < K; k0 += 32) {
        // issue all fragment loads for this K-step first (one clause, one wait)
        bf16x16 a0 = load_frag_a(ap0 + k0);
        bf16x16 a1 = load_frag_a(ap1 + k0);
        bf16x16 b0 = load_frag_b(bp0 + k0);
        bf16x16 b1 = load_frag_b(bp1 + k0);
        bf16x16 b2 = load_frag_b(bp2 + k0);
        bf16x16 b3 = load_frag_b(bp3 + k0);
        acc[0][0] = __builtin_amdgcn_wmma_f32_16x16x32_bf16(false, a0, false, b0, (short)0, acc[0][0], false, false);
        acc[0][1] = __builtin_amdgcn_wmma_f32_16x16x32_bf16(false, a0, false, b1, (short)0, acc[0][1], false, false);
        acc[0][2] = __builtin_amdgcn_wmma_f32_16x16x32_bf16(false, a0, false, b2, (short)0, acc[0][2], false, false);
        acc[0][3] = __builtin_amdgcn_wmma_f32_16x16x32_bf16(false, a0, false, b3, (short)0, acc[0][3], false, false);
        acc[1][0] = __builtin_amdgcn_wmma_f32_16x16x32_bf16(false, a1, false, b0, (short)0, acc[1][0], false, false);
        acc[1][1] = __builtin_amdgcn_wmma_f32_16x16x32_bf16(false, a1, false, b1, (short)0, acc[1][1], false, false);
        acc[1][2] = __builtin_amdgcn_wmma_f32_16x16x32_bf16(false, a1, false, b2, (short)0, acc[1][2], false, false);
        acc[1][3] = __builtin_amdgcn_wmma_f32_16x16x32_bf16(false, a1, false, b3, (short)0, acc[1][3], false, false);
    }

#pragma unroll
    for (int mi = 0; mi < 2; ++mi) {
#pragma unroll
        for (int t = 0; t < 4; ++t) {
            const int col = n0 + t * 16 + l;
#pragma unroll
            for (int r = 0; r < 8; ++r) {
                const int row = m0 + mi * 16 + r + hi * 8;
                float v = acc[mi][t][r];
                if (MODE == 0) {
                    if (bias_n) v += bias_n[col];
                    if (bias_m) v += bias_m[row];
                    ((bf16*)Cout)[c_base + (long long)z * c_batch + (long long)row * c_rs + col] = (bf16)v;
                } else if (MODE == 1) {
                    v += bias_n[col];
                    ((float*)Cout)[c_base + (long long)z * c_batch + (long long)row * c_rs + col] = v;
                } else {
                    const long long idx = (long long)z * c_batch + (long long)row * c_rs + col;
                    const float g = 1.0f / (1.0f + __expf(-gate_w[0]));
                    v = v * scale + g * align_bias[idx];
                    v = fmaxf(v, -FMAXF);
                    if (mask[idx]) v = -FMAXF; // (v - FMAX) clamped == -FMAX for finite logits
                    ((float*)Cout)[idx] = v;
                }
            }
        }
    }
}

// ---------------- elementwise casts ------------
__global__ __launch_bounds__(256) void cast_f32_bf16(const float* __restrict__ in,
                                                     bf16* __restrict__ out, long long n) {
    long long i = (long long)blockIdx.x * blockDim.x + threadIdx.x;
    const long long stride = (long long)gridDim.x * blockDim.x;
    for (; i < n; i += stride) out[i] = (bf16)in[i];
}

// Y (y,b,1024) fp32 -> concat rows (y*B+b, 1536) cols [0,1024)
__global__ __launch_bounds__(256) void cast_y_to_concat(const float* __restrict__ in,
                                                        bf16* __restrict__ out, long long n) {
    long long i = (long long)blockIdx.x * blockDim.x + threadIdx.x;
    const long long stride = (long long)gridDim.x * blockDim.x;
    for (; i < n; i += stride) {
        const long long m = i / YD, j = i % YD;
        out[m * CD + j] = (bf16)in[i];
    }
}

// ---------------- softmax over x (row length 2048), one block per (b,y) ------------
__global__ __launch_bounds__(256) void softmax_rows(const float* __restrict__ logits,
                                                    bf16* __restrict__ attn) {
    const long long base = (long long)blockIdx.x * XLEN;
    const int tid = threadIdx.x;
    float vals[8];
    float m = -FMAXF;
#pragma unroll
    for (int i = 0; i < 8; ++i) {
        vals[i] = logits[base + tid + i * 256];
        m = fmaxf(m, vals[i]);
    }
#pragma unroll
    for (int o = 16; o; o >>= 1) m = fmaxf(m, __shfl_xor(m, o, 32));
    __shared__ float redm[8];
    if ((tid & 31) == 0) redm[tid >> 5] = m;
    __syncthreads();
#pragma unroll
    for (int i = 0; i < 8; ++i) m = fmaxf(m, redm[i]);

    float s = 0.0f;
#pragma unroll
    for (int i = 0; i < 8; ++i) { vals[i] = __expf(vals[i] - m); s += vals[i]; }
#pragma unroll
    for (int o = 16; o; o >>= 1) s += __shfl_xor(s, o, 32);
    __shared__ float reds[8];
    if ((tid & 31) == 0) reds[tid >> 5] = s;
    __syncthreads();
    s = 0.0f;
#pragma unroll
    for (int i = 0; i < 8; ++i) s += reds[i];
    const float inv = 1.0f / s;
#pragma unroll
    for (int i = 0; i < 8; ++i) attn[base + tid + i * 256] = (bf16)(vals[i] * inv);
}

// ---------------- host ------------
extern "C" void kernel_launch(void* const* d_in, const int* in_sizes, int n_in,
                              void* d_out, int out_size, void* d_ws, size_t ws_size,
                              hipStream_t stream) {
    const float* X         = (const float*)d_in[0];
    const float* Y         = (const float*)d_in[1];
    const float* alignment = (const float*)d_in[2];
    const unsigned char* pad_mask = (const unsigned char*)d_in[3];
    const float* Xk_w = (const float*)d_in[4];
    const float* Xk_b = (const float*)d_in[5];
    const float* Xv_w = (const float*)d_in[6];
    const float* Xv_b = (const float*)d_in[7];
    const float* Yq_w = (const float*)d_in[8];
    const float* Yq_b = (const float*)d_in[9];
    const float* Yw_w = (const float*)d_in[10];
    const float* Yw_b = (const float*)d_in[11];
    const float* align_w = (const float*)d_in[12];
    float* out = (float*)d_out;

    char* ws = (char*)d_ws;
    size_t off = 0;
    auto alloc = [&](size_t bytes) -> void* {
        void* p = ws + off;
        off = (off + bytes + 255) & ~(size_t)255;
        return p;
    };
    bf16*  Xbf    = (bf16*)alloc((size_t)MBX * XD * 2);
    bf16*  Xkw    = (bf16*)alloc((size_t)HD * XD * 2);
    bf16*  Xvw    = (bf16*)alloc((size_t)HD * XD * 2);
    bf16*  Yqw    = (bf16*)alloc((size_t)HD * YD * 2);
    bf16*  Ywbf   = (bf16*)alloc((size_t)YOUT * CD * 2);
    bf16*  xk     = (bf16*)alloc((size_t)MBX * HD * 2);            // (x*B+b, hd)
    bf16*  yq     = (bf16*)alloc((size_t)MBY * HD * 2);            // (y*B+b, hd)
    bf16*  xvT    = (bf16*)alloc((size_t)BB * HD * XLEN * 2);      // [b][h][x]
    float* logits = (float*)alloc((size_t)BB * YLEN * XLEN * 4);   // [b][y][x]
    bf16*  attn   = (bf16*)alloc((size_t)BB * YLEN * XLEN * 2);    // [b][y][x]
    bf16*  concat = (bf16*)alloc((size_t)MBY * CD * 2);            // (y*B+b, 1536)

    const float scale = 0.04419417382415922f; // 1/sqrt(512)

    // ---- casts
    cast_f32_bf16<<<8192, 256, 0, stream>>>(X,    Xbf,  (long long)MBX * XD);
    cast_f32_bf16<<<2048, 256, 0, stream>>>(Xk_w, Xkw,  (long long)HD * XD);
    cast_f32_bf16<<<2048, 256, 0, stream>>>(Xv_w, Xvw,  (long long)HD * XD);
    cast_f32_bf16<<<2048, 256, 0, stream>>>(Yq_w, Yqw,  (long long)HD * YD);
    cast_f32_bf16<<<6144, 256, 0, stream>>>(Yw_w, Ywbf, (long long)YOUT * CD);
    cast_y_to_concat<<<4096, 256, 0, stream>>>(Y, concat, (long long)MBY * YD);

    // ---- xk = X @ Xk_w^T + bk : M=16384 N=512 K=1024 -> 512*8=4096 tiles / 8 waves
    gemm_wmma<0><<<dim3(512, 1, 1), 256, 0, stream>>>(
        Xbf, XD, 0, Xkw, XD, 0, Xk_b, nullptr,
        xk, 0, 0, HD, MBX, HD, XD, nullptr, nullptr, nullptr, 0.0f);

    // ---- yq = Y @ Yq_w^T + bq : M=8192 N=512 K=1024 (A = Y's bf16 copy inside concat)
    gemm_wmma<0><<<dim3(256, 1, 1), 256, 0, stream>>>(
        concat, CD, 0, Yqw, YD, 0, Yq_b, nullptr,
        yq, 0, 0, HD, MBY, HD, YD, nullptr, nullptr, nullptr, 0.0f);

    // ---- xvT[b][h][x] = Xv_w @ X_b^T + bv (per batch) : M=512(h) N=2048(x) K=1024
    //      16*32=512 tiles -> 64 blocks, z=8
    gemm_wmma<0><<<dim3(64, 1, BB), 256, 0, stream>>>(
        Xvw, XD, 0, Xbf, BB * XD, XD, nullptr, Xv_b,
        xvT, 0, (long long)HD * XLEN, XLEN, HD, XLEN, XD, nullptr, nullptr, nullptr, 0.0f);

    // ---- logits[b,y,x] = (yq_b @ xk_b^T)*scale + sigmoid(w)*align, mask, clamp
    //      M=1024(y) N=2048(x) K=512 : 32*32=1024 tiles -> 128 blocks, z=8
    gemm_wmma<2><<<dim3(128, 1, BB), 256, 0, stream>>>(
        yq, BB * HD, HD, xk, BB * HD, HD, nullptr, nullptr,
        logits, 0, (long long)YLEN * XLEN, XLEN, YLEN, XLEN, HD,
        alignment, pad_mask, align_w, scale);

    // ---- softmax over x, attn bf16
    softmax_rows<<<BB * YLEN, 256, 0, stream>>>(logits, attn);

    // ---- attn_feat -> concat cols [1024,1536) : M=1024(y) N=512(h) K=2048, per batch
    //      32*8=256 tiles -> 32 blocks, z=8
    gemm_wmma<0><<<dim3(32, 1, BB), 256, 0, stream>>>(
        attn, XLEN, (long long)YLEN * XLEN, xvT, XLEN, (long long)HD * XLEN,
        nullptr, nullptr,
        concat, /*c_base=*/YD, /*c_batch=*/CD, /*c_rs=*/BB * CD,
        YLEN, HD, XLEN, nullptr, nullptr, nullptr, 0.0f);

    // ---- out = concat @ Yw_w^T + bw : M=8192 N=1024 K=1536, f32 out
    //      256*16=4096 tiles -> 512 blocks
    gemm_wmma<1><<<dim3(512, 1, 1), 256, 0, stream>>>(
        concat, CD, 0, Ywbf, CD, 0, Yw_b, nullptr,
        out, 0, 0, YOUT, MBY, YOUT, CD, nullptr, nullptr, nullptr, 0.0f);
}
